// SparseSecondOrderAttention_46033459478882
// MI455X (gfx1250) — compile-verified, matrix operand order checked by reading, not run on hardware
//
#include <hip/hip_runtime.h>

// ---------------------------------------------------------------------------
// SparseSecondOrderAttention for MI455X (gfx1250, wave32, WMMA f16 16x16x32)
// B=1, C=2048, D=512, H=8, DH=64, SPARSE_K=128
// ---------------------------------------------------------------------------

typedef __attribute__((ext_vector_type(16))) _Float16 v16h;
typedef __attribute__((ext_vector_type(8)))  _Float16 v8h;
typedef __attribute__((ext_vector_type(8)))  float    v8f;

#define DEV __device__ __forceinline__

static constexpr int Cseq  = 2048;
static constexpr int Dmod  = 512;
static constexpr int Hh    = 8;
static constexpr int DH    = 64;
static constexpr int KTOP  = 128;
static constexpr int SROW  = 2052;   // padded LDS row stride (floats) for scores
static constexpr int QPTS  = 20;     // qpt row stride (floats), 16B aligned
// LDS layout (floats unless noted)
static constexpr int OFF_SCORES = 0;
static constexpr int OFF_QTILE  = OFF_SCORES + 16 * SROW;
static constexpr int OFF_WQK    = OFF_QTILE + 16 * 68;
static constexpr int OFF_QPT    = OFF_WQK + 64 * 65;
static constexpr int OFF_OUTL   = OFF_QPT + 64 * QPTS;
static constexpr int OFF_KTILE  = OFF_OUTL + 16 * 68;         // f16 [2][16*64]
static constexpr int SMEM_BYTES = OFF_KTILE * 4 + 2 * 16 * 64 * 2;

// ---- WMMA fragment helper (ISA 16-bit A/B layout: lane m=l&15, hb=l>>4,
//      element e -> K = (e&7) + 16*(e>>3) + 8*hb; two contiguous 8-half runs) ----
DEV v16h load_frag_f16(const _Float16* __restrict__ base, int ld) {
  const int l  = threadIdx.x & 31;
  const int m  = l & 15;
  const int hb = l >> 4;
  const _Float16* p = base + m * ld + hb * 8;
  v8h lo = *(const v8h*)(p);
  v8h hi = *(const v8h*)(p + 16);
  return __builtin_shufflevector(lo, hi, 0,1,2,3,4,5,6,7,8,9,10,11,12,13,14,15);
}

DEV float fast_tanh(float x) {
#if __has_builtin(__builtin_amdgcn_tanhf)
  return __builtin_amdgcn_tanhf(x);          // v_tanh_f32 (gfx1250 trans op)
#else
  float r;
  asm volatile("v_tanh_f32 %0, %1" : "=v"(r) : "v"(x));
  return r;
#endif
}

DEV float gelu_t(float x) {                  // tanh-form gelu (1 trans op)
  float u = 0.7978845608028654f * (x + 0.044715f * x * x * x);
  return 0.5f * x * (1.0f + fast_tanh(u));
}

// ---- CDNA5 Tensor Data Mover: stage a 16x64 f16 tile (row stride 512 halves)
//      from global into LDS. Guarded: falls back to pipelined global loads. ----
#if __has_builtin(__builtin_amdgcn_tensor_load_to_lds)
#define HAVE_TDM 1
#if __clang_major__ >= 23
#pragma message("TDM: using 6-arg __builtin_amdgcn_tensor_load_to_lds")
#else
#pragma message("TDM: using 5-arg __builtin_amdgcn_tensor_load_to_lds")
#endif
typedef __attribute__((ext_vector_type(4))) unsigned int v4u;
typedef __attribute__((ext_vector_type(4))) int v4i;
typedef __attribute__((ext_vector_type(8))) int v8i;

DEV void tdm_load_tile_16x64_f16(const _Float16* gsrc, unsigned lds_off) {
  unsigned long long ga = (unsigned long long)gsrc;
  v4u g0;
  g0[0] = 1u;                                    // count=1, user-mode D#
  g0[1] = lds_off;                               // lds_addr (bytes)
  g0[2] = (unsigned)(ga & 0xffffffffull);        // global_addr[31:0]
  g0[3] = (unsigned)((ga >> 32) & 0x01ffffffull) // global_addr[56:32]
        | (2u << 30);                            // type = 2 ("image")
  v8i g1;
  g1[0] = (int)(1u << 16);                       // data_size=1 (2 bytes/elem)
  g1[1] = (int)(64u << 16);                      // tensor_dim0 = 64
  g1[2] = (int)(2048u << 16);                    // tensor_dim1 = 2048
  g1[3] = (int)(64u << 16);                      // tile_dim0 = 64
  g1[4] = (int)(16u);                            // tile_dim1 = 16, tile_dim2 = 0
  g1[5] = (int)(512u);                           // tensor_dim0_stride = 512
  g1[6] = 0;                                     // tensor_dim1_stride = 0 (2D)
  g1[7] = 0;
  v4i z4 = {0, 0, 0, 0};
#if __clang_major__ >= 23
  v8i z8 = {0, 0, 0, 0, 0, 0, 0, 0};
  __builtin_amdgcn_tensor_load_to_lds(g0, g1, z4, z4, z8, 0);
#else
  __builtin_amdgcn_tensor_load_to_lds(g0, g1, z4, z4, 0);
#endif
}

DEV void wait_tensorcnt0() {
#if __has_builtin(__builtin_amdgcn_s_wait_tensorcnt)
  __builtin_amdgcn_s_wait_tensorcnt(0);
#else
  asm volatile("s_wait_tensorcnt 0x0" ::: "memory");
#endif
}
#else
#define HAVE_TDM 0
#pragma message("TDM: builtin NOT available; using global-load fallback")
#endif

// ---------------------------------------------------------------------------
// f32 -> f16 (with optional row slice) conversion
// ---------------------------------------------------------------------------
__global__ void cvt_slice_kernel(const float* __restrict__ src,
                                 _Float16* __restrict__ dst,
                                 int rows, int cols, int src_ld, int src_off) {
  int idx = blockIdx.x * blockDim.x + threadIdx.x;
  if (idx >= rows * cols) return;
  int r = idx / cols, c = idx - r * cols;
  dst[idx] = (_Float16)src[r * src_ld + src_off + c];
}

// ---------------------------------------------------------------------------
// Generic WMMA GEMM: C[M,N] = A[M,K] @ Bt[N,K]^T (+bias), optional f32/f16 and
// transposed outputs. Block = 4 waves, 64x64 tile; grid (M/64, N/64).
// Software-pipelined; unroll x2 so the rotation needs no register copies.
// ---------------------------------------------------------------------------
__global__ __launch_bounds__(128) void wmma_gemm_kernel(
    const _Float16* __restrict__ A, int lda,
    const _Float16* __restrict__ Bt, int ldb,
    const float* __restrict__ bias,
    float* __restrict__ C32, int ldc32,
    _Float16* __restrict__ C16, int ldc16,
    _Float16* __restrict__ Ct16, int ldt16,
    float* __restrict__ Ct32, int ldt32,
    int K) {
  const int wave = threadIdx.x >> 5;
  const int row0 = blockIdx.x * 64 + wave * 16;
  const int col0 = blockIdx.y * 64;
  const int nk = K >> 5;

  v8f acc[4] = {};
  v16h af = load_frag_f16(A + row0 * lda, lda);
  v16h bf[4];
#pragma unroll
  for (int n = 0; n < 4; ++n)
    bf[n] = load_frag_f16(Bt + (col0 + n * 16) * ldb, ldb);

#pragma clang loop unroll_count(2)
  for (int s = 0; s < nk - 1; ++s) {
    const int ks = (s + 1) * 32;
    v16h afn = load_frag_f16(A + row0 * lda + ks, lda);
    v16h bn[4];
#pragma unroll
    for (int n = 0; n < 4; ++n)
      bn[n] = load_frag_f16(Bt + (col0 + n * 16) * ldb + ks, ldb);
#pragma unroll
    for (int n = 0; n < 4; ++n)
      acc[n] = __builtin_amdgcn_wmma_f32_16x16x32_f16(
          false, af, false, bf[n], (short)0, acc[n], false, false);
    af = afn;
#pragma unroll
    for (int n = 0; n < 4; ++n) bf[n] = bn[n];
  }
#pragma unroll
  for (int n = 0; n < 4; ++n)
    acc[n] = __builtin_amdgcn_wmma_f32_16x16x32_f16(
        false, af, false, bf[n], (short)0, acc[n], false, false);

  const int l = threadIdx.x & 31;
  const int col = l & 15, r0 = (l >> 4) * 8;
#pragma unroll
  for (int n = 0; n < 4; ++n) {
    int c = col0 + n * 16 + col;
    float bvv = bias ? bias[c] : 0.0f;
#pragma unroll
    for (int i = 0; i < 8; ++i) {
      float v = acc[n][i] + bvv;
      int r = row0 + r0 + i;
      if (C32)  C32[r * ldc32 + c] = v;
      if (C16)  C16[r * ldc16 + c] = (_Float16)v;
      if (Ct16) Ct16[c * ldt16 + r] = (_Float16)v;
      if (Ct32) Ct32[c * ldt32 + r] = v;
    }
  }
}

// ---------------------------------------------------------------------------
// Fused second-order scores + exact top-k threshold + softmax + attn@v.
// Grid: (C/16, H). Block: 256 threads = 8 waves. Wave w owns d' = 8w..8w+7.
// ---------------------------------------------------------------------------
__global__ __launch_bounds__(256) void sparse_so_attn_kernel(
    const float*    __restrict__ qbuf,  // [2048][512] f32 (q, head-major cols)
    const _Float16* __restrict__ kh,    // [2048][512] f16
    const _Float16* __restrict__ vT,    // [512][2048] f16 (v transposed)
    const float*    __restrict__ qp,    // [2048][512] f32
    const float*    __restrict__ kpT,   // [512][2048] f32 (kp transposed)
    const float*    __restrict__ W1,    // [64][192]
    const float*    __restrict__ b1,    // [64]
    const float*    __restrict__ W2,    // [64]
    const float*    __restrict__ b2,    // [1]
    _Float16*       __restrict__ outh)  // [2048][512] f16
{
  extern __shared__ char smem_raw[];
  float* smf     = (float*)smem_raw;
  float* scores  = smf + OFF_SCORES;           // [16][SROW]
  float* qtile   = smf + OFF_QTILE;            // [16][68]
  float* wqk     = smf + OFF_WQK;              // [64][65]  wqk[d][d']=W1[d'][128+d]
  float* qpt     = smf + OFF_QPT;              // [64][QPTS] qpt[d'][q]=qp+b1
  float* outl    = smf + OFF_OUTL;             // [16][68]
  _Float16* ktile = (_Float16*)(smf + OFF_KTILE); // [2][16*64] f16 (TDM staging)
  (void)ktile;

  const int h    = blockIdx.y;
  const int q0   = blockIdx.x * 16;
  const int tid  = threadIdx.x;
  const int wave = tid >> 5;
  const int l    = tid & 31;
  const int m    = l & 15;
  const int hb   = l >> 4;
  const int NT   = Cseq / 16;
  const float scale = 0.125f;                  // 1/sqrt(64)

  // ---------------- stage ----------------
  float sinit = b2[0] * scale;
  for (int i = tid; i < 16 * SROW; i += 256) scores[i] = sinit;
  for (int i = tid; i < 16 * 68;  i += 256) outl[i] = 0.0f;
  for (int i = tid; i < 16 * 64;  i += 256) {
    int r = i >> 6, c = i & 63;
    qtile[r * 68 + c] = qbuf[(q0 + r) * Dmod + h * DH + c];
  }
  for (int i = tid; i < 64 * 64; i += 256) {
    int d = i >> 6, dp = i & 63;
    wqk[d * 65 + dp] = W1[dp * 192 + 128 + d];
  }
  for (int i = tid; i < 64 * 16; i += 256) {
    int dp = i >> 4, r = i & 15;
    qpt[dp * QPTS + r] = qp[(q0 + r) * Dmod + h * DH + dp] + b1[dp];
  }
#if HAVE_TDM
  if (wave == 0) {                       // preload key tile 0 into buffer 0
    tdm_load_tile_16x64_f16(kh + h * DH,
                            (unsigned)(unsigned long long)(const void*)ktile);
    wait_tensorcnt0();
  }
#endif
  __syncthreads();

  // ---- build register-resident A fragments: A_{d'}[q,d] = q[q,d]*wqk[d,d'] ----
  v16h afr[8][2];
#pragma unroll
  for (int j = 0; j < 8; ++j) {
    int dp = wave * 8 + j;
#pragma unroll
    for (int s = 0; s < 2; ++s) {
      v16h f;
#pragma unroll
      for (int e = 0; e < 16; ++e) {
        int K = s * 32 + (e & 7) + ((e >> 3) * 16) + hb * 8;
        f[e] = (_Float16)(qtile[m * 68 + K] * wqk[K * 65 + dp]);
      }
      afr[j][s] = f;
    }
  }
  float w2v[8];
#pragma unroll
  for (int j = 0; j < 8; ++j) w2v[j] = W2[wave * 8 + j];

  // ---------------- main loop over 128 key tiles (pipelined) ----------------
#if !HAVE_TDM
  v16h bf0 = load_frag_f16(kh + h * DH, Dmod);
  v16h bf1 = load_frag_f16(kh + h * DH + 32, Dmod);
#endif
  float kpv[8];
#pragma unroll
  for (int j = 0; j < 8; ++j)
    kpv[j] = kpT[(h * DH + wave * 8 + j) * Cseq + m];

#pragma clang loop unroll_count(2)
  for (int t = 0; t < NT; ++t) {
    const int kb  = t * 16;
    const int kb2 = (t + 1 < NT) ? kb + 16 : kb;   // next-tile prefetch index
#if HAVE_TDM
    if (wave == 0 && t + 1 < NT)                   // async-stage next tile
      tdm_load_tile_16x64_f16(
          kh + kb2 * Dmod + h * DH,
          (unsigned)(unsigned long long)(const void*)(ktile + ((t + 1) & 1) * 1024));
    v16h bf0 = load_frag_f16(ktile + (t & 1) * 1024, 64);
    v16h bf1 = load_frag_f16(ktile + (t & 1) * 1024 + 32, 64);
#else
    __builtin_prefetch(kh + kb2 * Dmod + h * DH, 0, 1);
    v16h b0n = load_frag_f16(kh + kb2 * Dmod + h * DH, Dmod);
    v16h b1n = load_frag_f16(kh + kb2 * Dmod + h * DH + 32, Dmod);
#endif
    float kpn[8];
#pragma unroll
    for (int j = 0; j < 8; ++j)
      kpn[j] = kpT[(h * DH + wave * 8 + j) * Cseq + kb2 + m];

    v8f sacc = {};
#pragma unroll
    for (int j = 0; j < 8; ++j) {
      int dp = wave * 8 + j;
      float4 qa = *(const float4*)(qpt + dp * QPTS + hb * 8);
      float4 qb = *(const float4*)(qpt + dp * QPTS + hb * 8 + 4);
      v8f cinit;
      cinit[0] = qa.x + kpv[j]; cinit[1] = qa.y + kpv[j];
      cinit[2] = qa.z + kpv[j]; cinit[3] = qa.w + kpv[j];
      cinit[4] = qb.x + kpv[j]; cinit[5] = qb.y + kpv[j];
      cinit[6] = qb.z + kpv[j]; cinit[7] = qb.w + kpv[j];
      v8f r = __builtin_amdgcn_wmma_f32_16x16x32_f16(
          false, afr[j][0], false, bf0, (short)0, cinit, false, false);
      r = __builtin_amdgcn_wmma_f32_16x16x32_f16(
          false, afr[j][1], false, bf1, (short)0, r, false, false);
#pragma unroll
      for (int i = 0; i < 8; ++i)
        sacc[i] += gelu_t(r[i]) * w2v[j];
    }
#pragma unroll
    for (int i = 0; i < 8; ++i)
      atomicAdd(&scores[(hb * 8 + i) * SROW + kb + m], sacc[i] * scale);

#if HAVE_TDM
    if (wave == 0) wait_tensorcnt0();
    __syncthreads();                     // next LDS tile ready, old reads done
#else
    bf0 = b0n; bf1 = b1n;
#endif
#pragma unroll
    for (int j = 0; j < 8; ++j) kpv[j] = kpn[j];
  }
#if !HAVE_TDM
  __syncthreads();
#endif

  // ------------- exact top-K threshold + masked softmax (2 rows/wave) -------
  for (int rr = 0; rr < 2; ++rr) {
    const int row = wave + rr * 8;
    float* srow = scores + row * SROW;
    float    rv[64];
    unsigned mv[64];
#pragma unroll
    for (int j = 0; j < 64; ++j) {
      float f = srow[l + 32 * j];
      rv[j] = f;
      unsigned u = __float_as_uint(f);
      mv[j] = (u & 0x80000000u) ? ~u : (u | 0x80000000u);   // order-preserving
    }
    unsigned T = 0u;                     // K-th largest via bitwise search
    for (int b = 31; b >= 0; --b) {
      unsigned T2 = T | (1u << b);
      int cnt = 0;
#pragma unroll
      for (int j = 0; j < 64; ++j) cnt += (mv[j] >= T2) ? 1 : 0;
      for (int off = 16; off > 0; off >>= 1) cnt += __shfl_xor(cnt, off, 32);
      if (cnt >= KTOP) T = T2;
    }
    float mx = -3.4e38f;
#pragma unroll
    for (int j = 0; j < 64; ++j) mx = fmaxf(mx, rv[j]);
    for (int off = 16; off > 0; off >>= 1) mx = fmaxf(mx, __shfl_xor(mx, off, 32));
    float sum = 0.0f;
#pragma unroll
    for (int j = 0; j < 64; ++j)
      sum += (mv[j] >= T) ? __expf(rv[j] - mx) : 0.0f;
    for (int off = 16; off > 0; off >>= 1) sum += __shfl_xor(sum, off, 32);
    float inv = 1.0f / sum;
#pragma unroll
    for (int j = 0; j < 64; ++j)
      srow[l + 32 * j] = (mv[j] >= T) ? __expf(rv[j] - mx) * inv : 0.0f;
  }
  __syncthreads();

  // -------- attn @ v : wave -> (ntile = w&3, kstep half = w>>2), pipelined --
  {
    const int nt = wave & 3;
    const int s0 = (wave >> 2) * 32;
    const _Float16* vbase = vT + (h * DH + nt * 16) * Cseq;
    v8f acc = {};
    v16h bfc = load_frag_f16(vbase + s0 * 32, Cseq);
    for (int s = 0; s < 32; ++s) {
      const int ks = (s0 + s) * 32;
      const int ksn = (s + 1 < 32) ? ks + 32 : ks;
      v16h bfn = load_frag_f16(vbase + ksn, Cseq);
      const float* p = scores + m * SROW + ks + hb * 8;
      float4 a0 = *(const float4*)(p);
      float4 a1 = *(const float4*)(p + 4);
      float4 a2 = *(const float4*)(p + 16);
      float4 a3 = *(const float4*)(p + 20);
      v16h af;
      af[0]=(_Float16)a0.x;  af[1]=(_Float16)a0.y;  af[2]=(_Float16)a0.z;  af[3]=(_Float16)a0.w;
      af[4]=(_Float16)a1.x;  af[5]=(_Float16)a1.y;  af[6]=(_Float16)a1.z;  af[7]=(_Float16)a1.w;
      af[8]=(_Float16)a2.x;  af[9]=(_Float16)a2.y;  af[10]=(_Float16)a2.z; af[11]=(_Float16)a2.w;
      af[12]=(_Float16)a3.x; af[13]=(_Float16)a3.y; af[14]=(_Float16)a3.z; af[15]=(_Float16)a3.w;
      acc = __builtin_amdgcn_wmma_f32_16x16x32_f16(
          false, af, false, bfc, (short)0, acc, false, false);
      bfc = bfn;
    }
#pragma unroll
    for (int i = 0; i < 8; ++i)
      atomicAdd(&outl[(hb * 8 + i) * 68 + nt * 16 + m], acc[i]);
  }
  __syncthreads();

  for (int i = tid; i < 16 * 64; i += 256) {
    int r = i >> 6, c = i & 63;
    outh[(q0 + r) * Dmod + h * DH + c] = (_Float16)outl[r * 68 + c];
  }
}

// ---------------------------------------------------------------------------
// Host side
// ---------------------------------------------------------------------------
extern "C" void kernel_launch(void* const* d_in, const int* in_sizes, int n_in,
                              void* d_out, int out_size, void* d_ws, size_t ws_size,
                              hipStream_t stream) {
  (void)in_sizes; (void)n_in; (void)out_size; (void)ws_size;
  const float* x  = (const float*)d_in[0];
  const float* Wq = (const float*)d_in[1];
  const float* bq = (const float*)d_in[2];
  const float* Wk = (const float*)d_in[3];
  const float* bk = (const float*)d_in[4];
  const float* Wv = (const float*)d_in[5];
  const float* bv = (const float*)d_in[6];
  const float* W1 = (const float*)d_in[7];
  const float* b1 = (const float*)d_in[8];
  const float* W2 = (const float*)d_in[9];
  const float* b2 = (const float*)d_in[10];
  const float* Wo = (const float*)d_in[11];
  const float* bo = (const float*)d_in[12];
  float* out = (float*)d_out;

  char* wp = (char*)d_ws;
  auto alloc = [&](size_t b) { char* p = wp; wp += (b + 255) & ~(size_t)255; return p; };
  _Float16* xh   = (_Float16*)alloc((size_t)Cseq * Dmod * 2);
  _Float16* wqh  = (_Float16*)alloc((size_t)Dmod * Dmod * 2);
  _Float16* wkh  = (_Float16*)alloc((size_t)Dmod * Dmod * 2);
  _Float16* wvh  = (_Float16*)alloc((size_t)Dmod * Dmod * 2);
  _Float16* woh  = (_Float16*)alloc((size_t)Dmod * Dmod * 2);
  _Float16* w1ah = (_Float16*)alloc((size_t)DH * DH * 2);
  _Float16* w1bh = (_Float16*)alloc((size_t)DH * DH * 2);
  float*    qbuf = (float*)   alloc((size_t)Cseq * Dmod * 4);
  _Float16* qh   = (_Float16*)alloc((size_t)Cseq * Dmod * 2);
  _Float16* kh   = (_Float16*)alloc((size_t)Cseq * Dmod * 2);
  _Float16* vT   = (_Float16*)alloc((size_t)Dmod * Cseq * 2);
  float*    qpb  = (float*)   alloc((size_t)Cseq * Dmod * 4);
  float*    kpT  = (float*)   alloc((size_t)Dmod * Cseq * 4);
  _Float16* outh = (_Float16*)alloc((size_t)Cseq * Dmod * 2);

  auto cvt = [&](const float* s, _Float16* d, int rows, int cols, int ld, int off) {
    int n = rows * cols;
    cvt_slice_kernel<<<(n + 255) / 256, 256, 0, stream>>>(s, d, rows, cols, ld, off);
  };
  cvt(x,  xh,  Cseq, Dmod, Dmod, 0);
  cvt(Wq, wqh, Dmod, Dmod, Dmod, 0);
  cvt(Wk, wkh, Dmod, Dmod, Dmod, 0);
  cvt(Wv, wvh, Dmod, Dmod, Dmod, 0);
  cvt(Wo, woh, Dmod, Dmod, Dmod, 0);
  cvt(W1, w1ah, DH, DH, 192, 0);
  cvt(W1, w1bh, DH, DH, 192, DH);

  auto gemm = [&](const _Float16* A, int lda, const _Float16* Bt, int ldb,
                  const float* bias, int M, int N, int K,
                  float* C32, int ldc32, _Float16* C16, int ldc16,
                  _Float16* Ct16, int ldt16, float* Ct32, int ldt32) {
    dim3 grid(M / 64, N / 64);
    wmma_gemm_kernel<<<grid, 128, 0, stream>>>(A, lda, Bt, ldb, bias,
                                               C32, ldc32, C16, ldc16,
                                               Ct16, ldt16, Ct32, ldt32, K);
  };
  // q = x@Wq.T + bq  (f32 for A-frag build, f16 for qp GEMM)
  gemm(xh, Dmod, wqh, Dmod, bq, Cseq, Dmod, Dmod,
       qbuf, Dmod, qh, Dmod, nullptr, 0, nullptr, 0);
  // k = x@Wk.T + bk  (f16, row-major per head)
  gemm(xh, Dmod, wkh, Dmod, bk, Cseq, Dmod, Dmod,
       nullptr, 0, kh, Dmod, nullptr, 0, nullptr, 0);
  // v = x@Wv.T + bv  (f16 transposed -> vT[512][2048])
  gemm(xh, Dmod, wvh, Dmod, bv, Cseq, Dmod, Dmod,
       nullptr, 0, nullptr, 0, vT, Cseq, nullptr, 0);
  // per-head qp = q@W1a.T (f32), kp = k@W1b.T (f32 transposed)
  for (int h = 0; h < Hh; ++h) {
    gemm(qh + h * DH, Dmod, w1ah, DH, nullptr, Cseq, DH, DH,
         qpb + h * DH, Dmod, nullptr, 0, nullptr, 0, nullptr, 0);
    gemm(kh + h * DH, Dmod, w1bh, DH, nullptr, Cseq, DH, DH,
         nullptr, 0, nullptr, 0, nullptr, 0, kpT + (size_t)h * DH * Cseq, Cseq);
  }

  // fused second-order attention
  hipFuncSetAttribute((const void*)sparse_so_attn_kernel,
                      hipFuncAttributeMaxDynamicSharedMemorySize, SMEM_BYTES);
  sparse_so_attn_kernel<<<dim3(Cseq / 16, Hh), 256, SMEM_BYTES, stream>>>(
      qbuf, kh, vT, qpb, kpT, W1, b1, W2, b2, outh);

  // final projection: out = outh @ Wo.T + bo
  gemm(outh, Dmod, woh, Dmod, bo, Cseq, Dmod, Dmod,
       out, Dmod, nullptr, 0, nullptr, 0, nullptr, 0);
}
